// MultiHeadSelfAttention_8263517078145
// MI455X (gfx1250) — matrix operand order repaired and checked by reading, not compiled, not run on hardware
//
#include <hip/hip_runtime.h>

// ---------------------------------------------------------------------------
// MHA: B=8, S=2048, E=512, H=8, D=64 — bf16 WMMA everywhere, f32 accumulate.
// Attention is reduction-free: S^T = K*Q^T, rowsum via ones-WMMA, no LDS.
// GEMM orientations chosen so every epilogue store is a contiguous vector.
// ---------------------------------------------------------------------------

typedef __bf16 bf16_t;
typedef __attribute__((ext_vector_type(16))) __bf16 v16bf;
typedef __attribute__((ext_vector_type(8)))  __bf16 v8bf;
typedef __attribute__((ext_vector_type(8)))  float  v8f;

constexpr int Bx = 8;
constexpr int Sx = 2048;
constexpr int Ex = 512;
constexpr int Hx = 8;
constexpr int Dx = 64;

__device__ __forceinline__ v16bf make_frag(v8bf lo, v8bf hi) {
    v16bf r;
#pragma unroll
    for (int i = 0; i < 8; ++i) { r[i] = lo[i]; r[i + 8] = hi[i]; }
    return r;
}

__device__ __forceinline__ v8f wmma_bf16(v16bf a, v16bf b, v8f c) {
    // (neg_a, A, neg_b, B, c_mod, C, reuse_a, reuse_b)
    return __builtin_amdgcn_wmma_f32_16x16x32_bf16(false, a, false, b,
                                                   (short)0, c, false, false);
}

// Exchange with partner lane (lane ^ 16) — one v_permlanex16_b32.
__device__ __forceinline__ float xor16f(float v) {
#if __has_builtin(__builtin_amdgcn_permlanex16)
    return __int_as_float(__builtin_amdgcn_permlanex16(
        __float_as_int(v), __float_as_int(v), 0x76543210, 0xfedcba98,
        false, false));
#else
    return __shfl_xor(v, 16, 32);
#endif
}

// ---------------------------------------------------------------------------
// Kernel 0a: transpose + convert 512x512 f32 weight to n-major bf16 (W^T).
// Kernel 0b: maskf = -1e9 * mask (additive softmax bias, f32).
// ---------------------------------------------------------------------------
__global__ void transpose_w_kernel(const float* __restrict__ w,
                                   bf16_t* __restrict__ wT) {
    int idx = blockIdx.x * blockDim.x + threadIdx.x;
    if (idx >= Ex * Ex) return;
    int kin  = idx / Ex;
    int nout = idx % Ex;
    wT[(size_t)nout * Ex + kin] = (bf16_t)w[idx];
}

__global__ void maskf_kernel(const int* __restrict__ mask,
                             float* __restrict__ maskf) {
    int idx = blockIdx.x * blockDim.x + threadIdx.x;
    if (idx >= Bx * Sx) return;
    maskf[idx] = -1e9f * (float)mask[idx];
}

// ---------------------------------------------------------------------------
// Weights-as-A GEMM (transposed orientation): C[m=feature, n=token].
// Wt: n-major bf16 (feature rows, contiguous k).  Acts: token rows, contig k.
// 32 features x 64 tokens per wave; 8 WMMAs per K-step of 32.
// MODE 0: acts f32 (X), out bf16 -> (B,H,S,D)   (Q, K projections)
// MODE 2: acts bf16 (ctx), out f32 row-major    (output projection)
// Epilogue: 8 consecutive features live in one lane -> vector stores.
// ---------------------------------------------------------------------------
template <int MODE>
__global__ __launch_bounds__(256)
void gemm_wA_kernel(const bf16_t* __restrict__ Wt, const void* __restrict__ Xv,
                    const float* __restrict__ bias, void* __restrict__ Outv) {
    const int wave = blockIdx.x * (blockDim.x >> 5) + (threadIdx.x >> 5);
    const int Mt = Ex / 32;            // 16 feature-blocks
    const int Nt = (Bx * Sx) / 64;     // 256 token-blocks
    if (wave >= Mt * Nt) return;
    const int tm = wave % Mt;          // consecutive waves share the token tile
    const int tn = wave / Mt;
    const int lane = threadIdx.x & 31;
    const int nl = lane & 15;
    const int hf = lane >> 4;

    const int f0 = tm * 32;            // feature base
    const int t0 = tn * 64;            // token base

    v8f acc[2][4] = {};
    for (int k0 = 0; k0 < Ex; k0 += 32) {
        v16bf a[2];                    // weight fragments
#pragma unroll
        for (int i = 0; i < 2; ++i) {
            const bf16_t* W = Wt + (size_t)(f0 + i * 16 + nl) * Ex + k0;
            a[i] = make_frag(*(const v8bf*)(W + 16 * hf),
                             *(const v8bf*)(W + 16 * hf + 8));
        }
#pragma unroll
        for (int t = 0; t < 4; ++t) {  // activation fragments (n-major)
            v16bf bfr;
            if constexpr (MODE == 0) {
                const float* X =
                    (const float*)Xv + (size_t)(t0 + t * 16 + nl) * Ex + k0 + 16 * hf;
                v8f lo = *(const v8f*)X;
                v8f hi = *(const v8f*)(X + 8);
#pragma unroll
                for (int e = 0; e < 8; ++e) {
                    bfr[e]     = (bf16_t)lo[e];
                    bfr[e + 8] = (bf16_t)hi[e];
                }
            } else {
                const v8bf* X = (const v8bf*)((const bf16_t*)Xv +
                                (size_t)(t0 + t * 16 + nl) * Ex + k0 + 16 * hf);
                bfr = make_frag(X[0], X[1]);
            }
#pragma unroll
            for (int i = 0; i < 2; ++i)
                acc[i][t] = wmma_bf16(a[i], bfr, acc[i][t]);
        }
    }

#pragma unroll
    for (int i = 0; i < 2; ++i) {
        const int fbase = f0 + i * 16 + 8 * hf;             // 8 consecutive features
        const v8f bc = *(const v8f*)(bias + fbase);
#pragma unroll
        for (int t = 0; t < 4; ++t) {
            const int token = t0 + t * 16 + nl;
            if constexpr (MODE == 0) {
                v8bf ov;
#pragma unroll
                for (int r = 0; r < 8; ++r) ov[r] = (bf16_t)(acc[i][t][r] + bc[r]);
                const int b = token / Sx, s = token % Sx;
                const int h = fbase / Dx, d = fbase % Dx;
                *(v8bf*)((bf16_t*)Outv +
                         (((size_t)b * Hx + h) * Sx + s) * Dx + d) = ov;
            } else {
                v8f ov;
#pragma unroll
                for (int r = 0; r < 8; ++r) ov[r] = acc[i][t][r] + bc[r];
                *(v8f*)((float*)Outv + (size_t)token * Ex + fbase) = ov;
            }
        }
    }
}

// ---------------------------------------------------------------------------
// V projection (activations-as-A): C[m=token, n=feature] -> Vt (B,H,D,S).
// The 8 acc VGPRs of a lane are 8 consecutive s positions -> v8bf stores.
// ---------------------------------------------------------------------------
__global__ __launch_bounds__(256)
void gemm_v_kernel(const float* __restrict__ X, const bf16_t* __restrict__ Wt,
                   const float* __restrict__ bias, bf16_t* __restrict__ Vt) {
    const int wave = blockIdx.x * (blockDim.x >> 5) + (threadIdx.x >> 5);
    const int Mt = (Bx * Sx) / 32;     // 512 token-blocks
    const int Nt = Ex / 64;            // 8 feature-blocks
    if (wave >= Mt * Nt) return;
    const int tn = wave % Nt;          // consecutive waves share token rows
    const int tm = wave / Nt;
    const int lane = threadIdx.x & 31;
    const int nl = lane & 15;
    const int hf = lane >> 4;

    const int row0 = tm * 32;
    const int col0 = tn * 64;

    v8f acc[2][4] = {};
    for (int k0 = 0; k0 < Ex; k0 += 32) {
        v16bf a[2];
#pragma unroll
        for (int i = 0; i < 2; ++i) {
            const float* A = X + (size_t)(row0 + i * 16 + nl) * Ex + k0;
            v8f lo = *(const v8f*)(A + 16 * hf);
            v8f hi = *(const v8f*)(A + 16 * hf + 8);
#pragma unroll
            for (int e = 0; e < 8; ++e) {
                a[i][e]     = (bf16_t)lo[e];
                a[i][e + 8] = (bf16_t)hi[e];
            }
        }
#pragma unroll
        for (int t = 0; t < 4; ++t) {
            const v8bf* bp = (const v8bf*)(Wt + (size_t)(col0 + t * 16 + nl) * Ex +
                                           k0 + 16 * hf);
            v16bf bfr = make_frag(bp[0], bp[1]);
#pragma unroll
            for (int i = 0; i < 2; ++i)
                acc[i][t] = wmma_bf16(a[i], bfr, acc[i][t]);
        }
    }

#pragma unroll
    for (int t = 0; t < 4; ++t) {
        const int col = col0 + t * 16 + nl;     // feature
        const float bc = bias[col];
        const int h = col / Dx, d = col % Dx;
#pragma unroll
        for (int i = 0; i < 2; ++i) {
            const int trow0 = row0 + i * 16 + 8 * hf;   // 8 consecutive tokens
            const int b = trow0 / Sx, s0 = trow0 % Sx;
            v8bf ov;
#pragma unroll
            for (int r = 0; r < 8; ++r) ov[r] = (bf16_t)(acc[i][t][r] + bc);
            *(v8bf*)(Vt + (((size_t)b * Hx + h) * Dx + d) * Sx + s0) = ov;
        }
    }
}

// ---------------------------------------------------------------------------
// Reduction-free attention.  One wave = 16 query rows of one (b,h).
// Q,K: (B,H,S,D) bf16.  Vt: (B,H,D,S) bf16.  maskf: (B,S) f32 = -1e9*mask.
// Per 32-key block: S^T = K*Q^T (4 WMMA), P = exp(S/8 + maskf),
// P^T B-fragment assembled with one lane^16 exchange per row,
// rowsum via ones-WMMA (1), O^T += V^T*P^T (4 WMMA).  9 WMMA / 32 keys.
// ---------------------------------------------------------------------------
__global__ __launch_bounds__(256)
void attn_kernel(const bf16_t* __restrict__ Q, const bf16_t* __restrict__ K,
                 const bf16_t* __restrict__ Vt, const float* __restrict__ maskf,
                 bf16_t* __restrict__ ctx) {
    const int wave  = blockIdx.x * (blockDim.x >> 5) + (threadIdx.x >> 5);
    const int numQT = Sx / 16;                 // 128
    const int bh = wave / numQT;
    const int qt = wave % numQT;
    if (bh >= Bx * Hx) return;
    const int b = bh / Hx;
    const int h = bh % Hx;

    const int lane = threadIdx.x & 31;
    const int nl = lane & 15;
    const int hf = lane >> 4;

    const bf16_t* Qb = Q  + (size_t)bh * Sx * Dx;
    const bf16_t* Kb = K  + (size_t)bh * Sx * Dx;
    const bf16_t* Vb = Vt + (size_t)bh * Dx * Sx;
    const float*  mb = maskf + (size_t)b * Sx;

    // Q as B operand (n = query, contiguous d), two contraction halves.
    v16bf qb[2];
#pragma unroll
    for (int f = 0; f < 2; ++f) {
        const v8bf* p =
            (const v8bf*)(Qb + (size_t)(qt * 16 + nl) * Dx + f * 32 + 8 * hf);
        qb[f] = make_frag(p[0], p[2]);
    }

    v16bf ones;
#pragma unroll
    for (int e = 0; e < 16; ++e) ones[e] = (bf16_t)1.0f;

    v8f o[4] = {};   // O^T accumulator: 64 d-rows x 16 q-cols (4 C tiles)
    v8f ls   = {};   // per-query softmax denominators (all rows identical)

    for (int kb = 0; kb < Sx; kb += 32) {
        // ---- S^T tiles: keys kb..kb+15 and kb+16..kb+31 -------------------
        v8f st0 = {}, st1 = {};
#pragma unroll
        for (int f = 0; f < 2; ++f) {
            const bf16_t* k0row = Kb + (size_t)(kb + nl) * Dx + f * 32;
            v16bf ka0 = make_frag(*(const v8bf*)(k0row + 8 * hf),
                                  *(const v8bf*)(k0row + 16 + 8 * hf));
            st0 = wmma_bf16(ka0, qb[f], st0);
            const bf16_t* k1row = Kb + (size_t)(kb + 16 + nl) * Dx + f * 32;
            v16bf ka1 = make_frag(*(const v8bf*)(k1row + 8 * hf),
                                  *(const v8bf*)(k1row + 16 + 8 * hf));
            st1 = wmma_bf16(ka1, qb[f], st1);
        }

        // ---- P = exp(S/8 + maskf); key index = kb + 8*hf + r (+16) --------
        const v8f mf0 = *(const v8f*)(mb + kb + 8 * hf);
        const v8f mf1 = *(const v8f*)(mb + kb + 16 + 8 * hf);
        float p0[8], p1[8];
#pragma unroll
        for (int r = 0; r < 8; ++r) {
            p0[r] = __expf(fmaf(st0[r], 0.125f, mf0[r]));
            p1[r] = __expf(fmaf(st1[r], 0.125f, mf1[r]));
        }

        // ---- assemble P^T B-fragment (one lane^16 exchange per row) -------
        v16bf pt;
#pragma unroll
        for (int r = 0; r < 8; ++r) {
            const float sel  = hf ? p1[r] : p0[r];   // this lane's half
            const float send = hf ? p0[r] : p1[r];   // partner needs this
            const float xs   = xor16f(send);
            pt[r]     = (bf16_t)(hf ? xs : sel);
            pt[r + 8] = (bf16_t)(hf ? sel : xs);
        }

        // ---- denominators and O^T accumulation ----------------------------
        ls = wmma_bf16(ones, pt, ls);
#pragma unroll
        for (int t = 0; t < 4; ++t) {
            const bf16_t* vrow = Vb + (size_t)(t * 16 + nl) * Sx + kb;
            v16bf va = make_frag(*(const v8bf*)(vrow + 16 * hf),
                                 *(const v8bf*)(vrow + 16 * hf + 8));
            o[t] = wmma_bf16(va, pt, o[t]);
        }
    }

    // ---- normalize + store ctx: 8 contiguous bf16 per lane per tile -------
    const float inv = 1.0f / ls[0];
    const int token = b * Sx + qt * 16 + nl;
#pragma unroll
    for (int t = 0; t < 4; ++t) {
        v8bf out;
#pragma unroll
        for (int r = 0; r < 8; ++r) out[r] = (bf16_t)(o[t][r] * inv);
        *(v8bf*)(ctx + (size_t)token * Ex + h * Dx + t * 16 + 8 * hf) = out;
    }
}

// ---------------------------------------------------------------------------
// Host side
// ---------------------------------------------------------------------------
extern "C" void kernel_launch(void* const* d_in, const int* in_sizes, int n_in,
                              void* d_out, int out_size, void* d_ws,
                              size_t ws_size, hipStream_t stream) {
    const float* value = (const float*)d_in[0];
    const float* key   = (const float*)d_in[1];
    const float* query = (const float*)d_in[2];
    const int*   mask  = (const int*)d_in[3];
    const float* wq = (const float*)d_in[4];
    const float* bq = (const float*)d_in[5];
    const float* wk = (const float*)d_in[6];
    const float* bk = (const float*)d_in[7];
    const float* wv = (const float*)d_in[8];
    const float* bv = (const float*)d_in[9];
    const float* wo = (const float*)d_in[10];
    const float* bo = (const float*)d_in[11];

    char* ws = (char*)d_ws;
    size_t off = 0;
    auto alloc = [&](size_t bytes) -> void* {
        void* p = ws + off;
        off = (off + bytes + 255) & ~(size_t)255;
        return p;
    };
    const size_t wBytes = (size_t)Ex * Ex * sizeof(bf16_t);
    const size_t tBytes = (size_t)Bx * Hx * Sx * Dx * sizeof(bf16_t);
    bf16_t* wqT   = (bf16_t*)alloc(wBytes);
    bf16_t* wkT   = (bf16_t*)alloc(wBytes);
    bf16_t* wvT   = (bf16_t*)alloc(wBytes);
    bf16_t* woT   = (bf16_t*)alloc(wBytes);
    float*  maskf = (float*) alloc((size_t)Bx * Sx * sizeof(float));
    bf16_t* Qbf   = (bf16_t*)alloc(tBytes);   // (B,H,S,D)
    bf16_t* Kbf   = (bf16_t*)alloc(tBytes);   // (B,H,S,D)
    bf16_t* Vtb   = (bf16_t*)alloc(tBytes);   // (B,H,D,S)
    bf16_t* ctx   = (bf16_t*)alloc(tBytes);   // (B*S, E)

    // 1) weight transposes (bf16, n-major) + mask bias
    {
        const int n = Ex * Ex, blk = 256, grd = (n + blk - 1) / blk;
        transpose_w_kernel<<<grd, blk, 0, stream>>>(wq, wqT);
        transpose_w_kernel<<<grd, blk, 0, stream>>>(wk, wkT);
        transpose_w_kernel<<<grd, blk, 0, stream>>>(wv, wvT);
        transpose_w_kernel<<<grd, blk, 0, stream>>>(wo, woT);
        const int nm = Bx * Sx;
        maskf_kernel<<<(nm + 255) / 256, 256, 0, stream>>>(mask, maskf);
    }

    // 2) projections (4096 waves each, 8 waves per block)
    const int gwaves = 4096;
    const int gblk   = 256;
    const int ggrd   = gwaves / (gblk / 32);           // 512
    gemm_wA_kernel<0><<<ggrd, gblk, 0, stream>>>(wqT, query, bq, Qbf);
    gemm_wA_kernel<0><<<ggrd, gblk, 0, stream>>>(wkT, key,   bk, Kbf);
    gemm_v_kernel<<<ggrd, gblk, 0, stream>>>(value, wvT, bv, Vtb);

    // 3) attention
    {
        const int waves = Bx * Hx * (Sx / 16);         // 8192
        const int blk = 256;
        const int grd = waves / (blk / 32);            // 1024
        attn_kernel<<<grd, blk, 0, stream>>>(Qbf, Kbf, Vtb, maskf, ctx);
    }

    // 4) output projection -> f32
    gemm_wA_kernel<2><<<ggrd, gblk, 0, stream>>>(woT, ctx, bo, (float*)d_out);
}
